// MoD_15573551415345
// MI455X (gfx1250) — matrix operand order repaired and checked by reading, not compiled, not run on hardware
//
#include <hip/hip_runtime.h>
#include <cstdint>
#include <cstddef>

// ---------------- problem constants ----------------
#define BB 4
#define TT 4096
#define CC 1024
#define HH 4096
#define KTOP 512
#define BT (BB * TT)          // 16384
#define BK_ROWS (BB * KTOP)   // 2048

typedef __attribute__((ext_vector_type(16))) __bf16 v16bf;
typedef __attribute__((ext_vector_type(8)))  float  v8f;
typedef __attribute__((ext_vector_type(4)))  int    i32x4;

// ---------------- helpers ----------------
__device__ __forceinline__ uint16_t f2bf(float f) {
    uint32_t u = __float_as_uint(f);
    uint32_t r = u + 0x7fffu + ((u >> 16) & 1u);   // RNE
    return (uint16_t)(r >> 16);
}

// branch-free tanh-approx gelu (matches jax.nn.gelu approximate=True):
// tanh(u) = 1 - 2/(exp(2u)+1); exp->v_exp_f32, rcp->v_rcp_f32, no EXEC diamonds
__device__ __forceinline__ float gelu_tanh(float x) {
    float x3 = x * x * x;
    float u = 0.7978845608028654f * (x + 0.044715f * x3);
    float e = __expf(2.0f * u);
    float t = 1.0f - 2.0f * __builtin_amdgcn_rcpf(e + 1.0f);
    return 0.5f * x * (1.0f + t);
}

// 16-byte global->LDS copy: async DMA (ASYNCcnt) when the gfx1250 builtin exists,
// else synchronous VGPR round-trip.
__device__ __forceinline__ void copy16_g2l(uint16_t* lds_dst, const uint16_t* gsrc) {
#if defined(__HIP_DEVICE_COMPILE__) && __has_builtin(__builtin_amdgcn_global_load_async_to_lds_b128)
    __builtin_amdgcn_global_load_async_to_lds_b128(
        (__attribute__((address_space(1))) i32x4*)gsrc,
        (__attribute__((address_space(3))) i32x4*)lds_dst,
        0, 0);
#else
    *(uint4*)lds_dst = *(const uint4*)gsrc;
#endif
}

__device__ __forceinline__ void wait_async0() {
#if defined(__HIP_DEVICE_COMPILE__) && __has_builtin(__builtin_amdgcn_global_load_async_to_lds_b128) && __has_builtin(__builtin_amdgcn_s_wait_asynccnt)
    __builtin_amdgcn_s_wait_asynccnt(0);
#endif
}

// A-fragment: lane r=lane&15 is row M=r; half h: K = h*8 + {0..7} and 16+h*8+{0..7}
__device__ __forceinline__ v16bf load_frag_a(const uint16_t* s, int r0, int lane) {
    int r = lane & 15, h = lane >> 4;
    const uint16_t* p = s + (r0 + r) * 40 + h * 8;
    union { v16bf v; uint4 q[2]; } u;
    u.q[0] = *(const uint4*)(p);
    u.q[1] = *(const uint4*)(p + 16);
    return u.v;
}

// B-fragment from [n][k] image: column n = lane&15; half h: K = h*8 + {0..7} and 16+h*8+{0..7}
__device__ __forceinline__ v16bf load_frag_b(const uint16_t* s, int n0, int lane) {
    int r = lane & 15, h = lane >> 4;
    const uint16_t* p = s + (n0 + r) * 40 + h * 8;
    union { v16bf v; uint4 q[2]; } u;
    u.q[0] = *(const uint4*)(p);
    u.q[1] = *(const uint4*)(p + 16);
    return u.v;
}

// ---------------- small kernels ----------------
__global__ void init_flags_kernel(unsigned* anyflag) {
    int i = threadIdx.x;
    if (i < TT / 32) anyflag[i] = 0u;
}

// Transposing fp32 -> bf16 conversion through an LDS tile (both sides coalesced).
// src: R x N row-major fp32 ; dst: N x R row-major bf16 (dst[n][r] = src[r][n])
__global__ void transpose_convert_kernel(const float* __restrict__ src,
                                         uint16_t* __restrict__ dst,
                                         int R, int N) {
    __shared__ uint16_t tile[32][33];
    int tx = threadIdx.x & 31, ty = threadIdx.x >> 5;   // 32 x 8
    int c0 = blockIdx.x * 32, r0 = blockIdx.y * 32;
#pragma unroll
    for (int i = 0; i < 32; i += 8)
        tile[ty + i][tx] = f2bf(src[(size_t)(r0 + ty + i) * N + c0 + tx]);
    __syncthreads();
#pragma unroll
    for (int i = 0; i < 32; i += 8)
        dst[(size_t)(c0 + ty + i) * R + r0 + tx] = tile[tx][ty + i];
}

// Fused: out = x (copy) + router logits (x@w_mod) + aux logits (x@w_mlp). One pass over x.
__global__ void router_copy_kernel(const float* __restrict__ x,
                                   const float* __restrict__ wmod,
                                   const float* __restrict__ wmlp,
                                   float* __restrict__ out,
                                   float* __restrict__ rlog,
                                   float* __restrict__ mlog) {
    int tid = threadIdx.x;
    int lane = tid & 31, wave = tid >> 5;
    int token = blockIdx.x * 8 + wave;              // blockDim = 256 -> 8 waves
    const float4* xv = (const float4*)(x + (size_t)token * CC);
    float4* ov = (float4*)(out + (size_t)token * CC);
    const float4* m0 = (const float4*)wmod;
    const float4* m1 = (const float4*)wmlp;
    float s0 = 0.f, s1 = 0.f;
    for (int c = lane; c < CC / 4; c += 32) {
        float4 xx = xv[c];
        float4 a = m0[c];
        float4 b = m1[c];
        ov[c] = xx;
        s0 += xx.x * a.x + xx.y * a.y + xx.z * a.z + xx.w * a.w;
        s1 += xx.x * b.x + xx.y * b.y + xx.z * b.z + xx.w * b.w;
    }
    for (int m = 16; m; m >>= 1) { s0 += __shfl_xor(s0, m, 32); s1 += __shfl_xor(s1, m, 32); }
    if (lane == 0) { rlog[token] = s0; mlog[token] = s1; }
}

// Exact top-512 per batch via binary search on composite 64-bit keys:
// key = sortable(val)<<32 | (T-1-t). Matches jax.lax.top_k tie-breaking (lower
// index first). Prefix-sum compaction emits sel[] ascending by token position.
__global__ void __launch_bounds__(1024)
select_kernel(const float* __restrict__ rlog, int* __restrict__ sel,
              float* __restrict__ selw, unsigned* __restrict__ anyflag) {
    __shared__ unsigned s_total;
    __shared__ unsigned s_warp[32];
    int b = blockIdx.x;
    int tid = threadIdx.x;
    int lane = tid & 31, wave = tid >> 5;

    uint64_t key[4];
#pragma unroll
    for (int q = 0; q < 4; ++q) {
        int t = tid * 4 + q;
        uint32_t u = __float_as_uint(rlog[b * TT + t]);
        uint32_t s = (u & 0x80000000u) ? ~u : (u | 0x80000000u);   // monotonic map
        key[q] = ((uint64_t)s << 32) | (uint32_t)(TT - 1 - t);
    }

    uint64_t lo = 0, hi = ~0ull;
    while (lo < hi) {
        uint64_t mid = lo + ((hi - lo) >> 1) + 1;
        unsigned c = 0;
#pragma unroll
        for (int q = 0; q < 4; ++q) c += (key[q] >= mid) ? 1u : 0u;
        for (int m = 16; m; m >>= 1) c += __shfl_xor(c, m, 32);
        if (tid == 0) s_total = 0;
        __syncthreads();
        if (lane == 0) atomicAdd(&s_total, c);
        __syncthreads();
        unsigned total = s_total;
        __syncthreads();
        if (total >= KTOP) lo = mid; else hi = mid - 1;
    }
    uint64_t thresh = lo;   // exactly KTOP keys >= thresh (keys distinct)

    unsigned cnt = 0;
#pragma unroll
    for (int q = 0; q < 4; ++q) cnt += (key[q] >= thresh) ? 1u : 0u;

    unsigned incl = cnt;
    for (int d = 1; d < 32; d <<= 1) {
        unsigned t = __shfl_up(incl, d, 32);
        if ((int)lane >= d) incl += t;
    }
    if (lane == 31) s_warp[wave] = incl;
    __syncthreads();
    if (tid == 0) {
        unsigned a = 0;
        for (int i = 0; i < 32; ++i) { unsigned t = s_warp[i]; s_warp[i] = a; a += t; }
    }
    __syncthreads();
    unsigned pos = s_warp[wave] + incl - cnt;

#pragma unroll
    for (int q = 0; q < 4; ++q) {
        if (key[q] >= thresh) {
            int t = tid * 4 + q;
            sel[b * KTOP + pos] = t;
            selw[b * KTOP + pos] = rlog[b * TT + t];
            atomicOr(&anyflag[t >> 5], 1u << (t & 31));
            pos++;
        }
    }
}

__global__ void __launch_bounds__(1024)
aux_kernel(const float* __restrict__ mlog, const unsigned* __restrict__ anyflag,
           float* __restrict__ aux_out) {
    __shared__ float sw[32];
    int tid = threadIdx.x;
    float s = 0.f;
    for (int i = tid; i < BT; i += 1024) {
        float l = mlog[i];
        float tgt = (i < TT && ((anyflag[i >> 5] >> (i & 31)) & 1u)) ? 1.f : 0.f;
        float sp = fmaxf(l, 0.f) + log1pf(expf(-fabsf(l)));   // stable softplus
        s += sp - l * tgt;
    }
    for (int m = 16; m; m >>= 1) s += __shfl_xor(s, m, 32);
    if ((tid & 31) == 0) sw[tid >> 5] = s;
    __syncthreads();
    if (tid < 32) {
        float v = sw[tid];
        for (int m = 16; m; m >>= 1) v += __shfl_xor(v, m, 32);
        if (tid == 0) *aux_out = v / (float)BT;
    }
}

__global__ void gather_kernel(const float* __restrict__ x, const int* __restrict__ sel,
                              uint16_t* __restrict__ tokg) {
    int r = blockIdx.x;         // 2048 rows
    int tid = threadIdx.x;      // 256 threads * 4 floats = 1024 = C
    int b = r >> 9;
    int t = sel[r];
    float4 v = ((const float4*)(x + ((size_t)b * TT + t) * CC))[tid];
    uint16_t o[4] = { f2bf(v.x), f2bf(v.y), f2bf(v.z), f2bf(v.w) };
    *(uint2*)(tokg + (size_t)r * CC + tid * 4) = *(uint2*)o;
}

// ---------------- WMMA GEMMs ----------------
// Block tile 128x128, 256 threads = 8 waves (2m x 4n), wave tile 64x32.
// Both A and Bt are row-major with row stride K, so staging is uniform:
// async global->LDS b128 copies, double-buffered so the DMA of tile k+1
// overlaps the 8 v_wmma_f32_16x16x32_bf16 of tile k.

#define LDT 40   // LDS row stride (elements), padded; keeps 16B alignment

__device__ __forceinline__ void stage_tiles(uint16_t* As, uint16_t* Bs,
                                            const uint16_t* A, const uint16_t* Bt,
                                            int row0, int col0, int K, int kk, int tid) {
    int m = tid >> 1, kc = (tid & 1) << 4;
    const uint16_t* asrc = A  + (size_t)(row0 + m) * K + kk + kc;
    const uint16_t* bsrc = Bt + (size_t)(col0 + m) * K + kk + kc;
    copy16_g2l(As + m * LDT + kc,     asrc);
    copy16_g2l(As + m * LDT + kc + 8, asrc + 8);
    copy16_g2l(Bs + m * LDT + kc,     bsrc);
    copy16_g2l(Bs + m * LDT + kc + 8, bsrc + 8);
}

// GEMM1: h = gelu(tokg[2048x1024] @ w1) with w1 pre-transposed (w1T: HH x CC bf16)
__global__ void __launch_bounds__(256)
gemm1_kernel(const uint16_t* __restrict__ A, const uint16_t* __restrict__ Bt,
             uint16_t* __restrict__ Hout) {
    const int K = CC, N = HH;
    __shared__ uint16_t As[2][128 * LDT];
    __shared__ uint16_t Bs[2][128 * LDT];

    int tid = threadIdx.x;
    int lane = tid & 31, wave = tid >> 5;
    int wm = wave >> 2, wn = wave & 3;
    int row0 = blockIdx.x * 128, col0 = blockIdx.y * 128;

    v8f acc[4][2];
    const v8f vzero = {0.f, 0.f, 0.f, 0.f, 0.f, 0.f, 0.f, 0.f};
#pragma unroll
    for (int i = 0; i < 4; ++i)
#pragma unroll
        for (int j = 0; j < 2; ++j) acc[i][j] = vzero;

    stage_tiles(As[0], Bs[0], A, Bt, row0, col0, K, 0, tid);
    wait_async0();
    __syncthreads();

    for (int kk = 0; kk < K; kk += 32) {
        int cur = (kk >> 5) & 1;
        if (kk + 32 < K)
            stage_tiles(As[cur ^ 1], Bs[cur ^ 1], A, Bt, row0, col0, K, kk + 32, tid);

        v16bf af[4], bfr[2];
#pragma unroll
        for (int i = 0; i < 4; ++i) af[i] = load_frag_a(As[cur], wm * 64 + i * 16, lane);
#pragma unroll
        for (int j = 0; j < 2; ++j) bfr[j] = load_frag_b(Bs[cur], wn * 32 + j * 16, lane);
#pragma unroll
        for (int i = 0; i < 4; ++i)
#pragma unroll
            for (int j = 0; j < 2; ++j)
                acc[i][j] = __builtin_amdgcn_wmma_f32_16x16x32_bf16(
                    false, af[i], false, bfr[j], (short)0, acc[i][j], false, false);

        wait_async0();
        __syncthreads();
    }

    int ln = lane & 15, h8 = (lane >> 4) * 8;
#pragma unroll
    for (int i = 0; i < 4; ++i)
#pragma unroll
        for (int j = 0; j < 2; ++j) {
            int gm = row0 + wm * 64 + i * 16 + h8;
            int gn = col0 + wn * 32 + j * 16 + ln;
#pragma unroll
            for (int r = 0; r < 8; ++r)
                Hout[(size_t)(gm + r) * N + gn] = f2bf(gelu_tanh(acc[i][j][r]));
        }
}

// GEMM2: ffn = h[2048x4096] @ w2 with w2 pre-transposed (w2T: CC x HH bf16);
// epilogue scatters out[b, sel[row], col] += selw[row]*ffn (positions unique).
__global__ void __launch_bounds__(256)
gemm2_kernel(const uint16_t* __restrict__ A, const uint16_t* __restrict__ Bt,
             const int* __restrict__ sel, const float* __restrict__ selw,
             float* __restrict__ out) {
    const int K = HH;
    __shared__ uint16_t As[2][128 * LDT];
    __shared__ uint16_t Bs[2][128 * LDT];

    int tid = threadIdx.x;
    int lane = tid & 31, wave = tid >> 5;
    int wm = wave >> 2, wn = wave & 3;
    int row0 = blockIdx.x * 128, col0 = blockIdx.y * 128;

    v8f acc[4][2];
    const v8f vzero = {0.f, 0.f, 0.f, 0.f, 0.f, 0.f, 0.f, 0.f};
#pragma unroll
    for (int i = 0; i < 4; ++i)
#pragma unroll
        for (int j = 0; j < 2; ++j) acc[i][j] = vzero;

    stage_tiles(As[0], Bs[0], A, Bt, row0, col0, K, 0, tid);
    wait_async0();
    __syncthreads();

    for (int kk = 0; kk < K; kk += 32) {
        int cur = (kk >> 5) & 1;
        if (kk + 32 < K)
            stage_tiles(As[cur ^ 1], Bs[cur ^ 1], A, Bt, row0, col0, K, kk + 32, tid);

        v16bf af[4], bfr[2];
#pragma unroll
        for (int i = 0; i < 4; ++i) af[i] = load_frag_a(As[cur], wm * 64 + i * 16, lane);
#pragma unroll
        for (int j = 0; j < 2; ++j) bfr[j] = load_frag_b(Bs[cur], wn * 32 + j * 16, lane);
#pragma unroll
        for (int i = 0; i < 4; ++i)
#pragma unroll
            for (int j = 0; j < 2; ++j)
                acc[i][j] = __builtin_amdgcn_wmma_f32_16x16x32_bf16(
                    false, af[i], false, bfr[j], (short)0, acc[i][j], false, false);

        wait_async0();
        __syncthreads();
    }

    int ln = lane & 15, h8 = (lane >> 4) * 8;
#pragma unroll
    for (int i = 0; i < 4; ++i) {
#pragma unroll
        for (int r = 0; r < 8; ++r) {
            int grow = row0 + wm * 64 + i * 16 + h8 + r;   // global slot in [0, 2048)
            int b = grow >> 9;
            int t = sel[grow];
            float w = selw[grow];
            size_t rowbase = ((size_t)b * TT + t) * CC;
#pragma unroll
            for (int j = 0; j < 2; ++j) {
                size_t o = rowbase + (col0 + wn * 32 + j * 16 + ln);
                out[o] += w * acc[i][j][r];
            }
        }
    }
}

// ---------------- host launcher ----------------
extern "C" void kernel_launch(void* const* d_in, const int* in_sizes, int n_in,
                              void* d_out, int out_size, void* d_ws, size_t ws_size,
                              hipStream_t stream) {
    const float* x    = (const float*)d_in[0];   // B*T*C
    const float* wmod = (const float*)d_in[1];   // C
    const float* wmlp = (const float*)d_in[2];   // C
    const float* w1   = (const float*)d_in[3];   // C*H
    const float* w2   = (const float*)d_in[4];   // H*C

    float* out = (float*)d_out;                               // B*T*C floats
    float* aux = out + (size_t)BB * TT * CC;                  // +1 scalar

    char* ws = (char*)d_ws;
    size_t off = 0;
    uint16_t* w1t  = (uint16_t*)(ws + off); off += (size_t)CC * HH * 2;        // HHxCC bf16
    uint16_t* w2t  = (uint16_t*)(ws + off); off += (size_t)HH * CC * 2;        // CCxHH bf16
    uint16_t* tokg = (uint16_t*)(ws + off); off += (size_t)BK_ROWS * CC * 2;
    uint16_t* hbuf = (uint16_t*)(ws + off); off += (size_t)BK_ROWS * HH * 2;
    float*    rlog = (float*)(ws + off);    off += (size_t)BT * 4;
    float*    mlog = (float*)(ws + off);    off += (size_t)BT * 4;
    int*      sel  = (int*)(ws + off);      off += (size_t)BK_ROWS * 4;
    float*    selw = (float*)(ws + off);    off += (size_t)BK_ROWS * 4;
    unsigned* anyf = (unsigned*)(ws + off); off += (TT / 32) * 4;
    (void)ws_size; (void)in_sizes; (void)n_in; (void)out_size;

    init_flags_kernel<<<1, 128, 0, stream>>>(anyf);
    // w1 (CC x HH) -> w1t (HH x CC); w2 (HH x CC) -> w2t (CC x HH)
    transpose_convert_kernel<<<dim3(HH / 32, CC / 32), 256, 0, stream>>>(w1, w1t, CC, HH);
    transpose_convert_kernel<<<dim3(CC / 32, HH / 32), 256, 0, stream>>>(w2, w2t, HH, CC);
    router_copy_kernel<<<BT / 8, 256, 0, stream>>>(x, wmod, wmlp, out, rlog, mlog);
    select_kernel<<<BB, 1024, 0, stream>>>(rlog, sel, selw, anyf);
    aux_kernel<<<1, 1024, 0, stream>>>(mlog, anyf, aux);
    gather_kernel<<<BK_ROWS, 256, 0, stream>>>(x, sel, tokg);
    gemm1_kernel<<<dim3(BK_ROWS / 128, HH / 128), 256, 0, stream>>>(tokg, w1t, hbuf);
    gemm2_kernel<<<dim3(BK_ROWS / 128, CC / 128), 256, 0, stream>>>(hbuf, w2t, sel, selw, out);
}
